// VMamba_Fuse_BASE_34626026340766
// MI455X (gfx1250) — matrix operand order repaired and checked by reading, not compiled, not run on hardware
//
#include <hip/hip_runtime.h>
#include <hip/hip_bf16.h>
#include <math.h>

typedef __attribute__((ext_vector_type(16))) _Float16 v16h;
typedef __attribute__((ext_vector_type(8)))  float    v8f;

#define BSZ 2

__device__ __forceinline__ float siluf(float x){ return x / (1.f + __expf(-x)); }
__device__ __forceinline__ float geluf(float x){ return 0.5f*x*(1.f + erff(x*0.70710678118654752f)); }
__device__ __forceinline__ float softplusf(float x){ return (x > 20.f) ? x : log1pf(__expf(x)); }

// ---- weight convert fp32 [N,K] -> f16 [Npad,Kpad], zero-padded ----------------------
__global__ void k_cvt(const float* __restrict__ src, _Float16* __restrict__ dst,
                      int N, int K, int Npad, int Kpad) {
  int i = blockIdx.x*blockDim.x + threadIdx.x;
  if (i >= Npad*Kpad) return;
  int n = i / Kpad, k = i % Kpad;
  dst[i] = (n < N && k < K) ? (_Float16)src[(size_t)n*K + k] : (_Float16)0.f;
}

// ---- WMMA GEMM: C[M,N] = act(A[M,K] * W[N,K]^T + bias) + add ------------------------
// One wave computes a 16 x (16*NT) strip: one A fragment feeds NT WMMAs per K-chunk.
// Requirements: M%16==0, N%(16*NT)==0 (grid), K%32==0 (template). No guards.
// A fp32 row-major (lda, 8B-aligned rows), W f16 row-major [N,Kpad>=K] (ldw%32==0).
// Contiguous k order (wave32): lane half h holds k = 16h..16h+15 of A and B.
template<int ACT, bool HB, bool HA, int K, int NT>
__global__ void __launch_bounds__(32)
k_gemm(const float* __restrict__ A, const _Float16* __restrict__ W,
       const float* __restrict__ bias, const float* __restrict__ add,
       float* __restrict__ C, int lda, int ldw, int ldc, int addld) {
  int tm = blockIdx.x, tn = blockIdx.y;
  int lane = threadIdx.x & 31;
  int half = lane >> 4;
  int mr   = lane & 15;
  int ncol0 = tn*NT*16 + mr;                     // first output column for this lane
  const float*    Arow  = A + (size_t)(tm*16 + mr)*lda;
  const _Float16* Wbase = W + (size_t)ncol0*ldw + half*16;
  v8f acc[NT];
  #pragma unroll
  for (int nt = 0; nt < NT; ++nt) acc[nt] = {};
  #pragma unroll
  for (int k0 = 0; k0 < K; k0 += 32) {
    v16h a;
    #pragma unroll
    for (int v = 0; v < 8; ++v) {               // A: VGPR v holds K pair (b64 loads)
      int kb = k0 + half*16 + v*2;
      float2 a2 = *reinterpret_cast<const float2*>(Arow + kb);
      a[2*v]   = (_Float16)a2.x;
      a[2*v+1] = (_Float16)a2.y;
    }
    #pragma unroll
    for (int nt = 0; nt < NT; ++nt) {           // NT B strips share one A fragment
      v16h b = *reinterpret_cast<const v16h*>(Wbase + (size_t)nt*16*ldw + k0);
      acc[nt] = __builtin_amdgcn_wmma_f32_16x16x32_f16(false, a, false, b, (short)0,
                                                       acc[nt], false, false);
    }
  }
  float bv[NT];
  #pragma unroll
  for (int nt = 0; nt < NT; ++nt) bv[nt] = HB ? bias[ncol0 + nt*16] : 0.f;
  #pragma unroll
  for (int v = 0; v < 8; ++v) {                 // D: lane half*8+v rows
    int mrow = tm*16 + half*8 + v;
    float*       Crow = C + (size_t)mrow*ldc + ncol0;
    const float* Srow = HA ? (add + (size_t)mrow*addld + ncol0) : nullptr;
    #pragma unroll
    for (int nt = 0; nt < NT; ++nt) {
      float val = acc[nt][v] + bv[nt];
      if (ACT == 1) val = geluf(val);
      if (HA) val += Srow[nt*16];
      Crow[nt*16] = val;
    }
  }
}

// ---- dt projection (K=4) + bias + softplus: plain VALU kernel ----------------------
// XD: [4][R][48] (cols 0..3 = dts). dtw: [4][128][4]. DTB: [4][R][128].
__global__ void k_dt(const float* __restrict__ XD, const float* __restrict__ dtw,
                     const float* __restrict__ dtb, float* __restrict__ DTB, int R) {
  int idx = blockIdx.x*blockDim.x + threadIdx.x;
  if (idx >= 4*R*128) return;
  int d = idx & 127;
  int rk = idx >> 7;
  int r = rk % R;
  int k = rk / R;
  const float* xr = XD + ((size_t)k*R + r)*48;
  const float* wr = dtw + ((size_t)(k*128 + d))*4;
  float s = dtb[k*128 + d];
  #pragma unroll
  for (int j = 0; j < 4; ++j) s += xr[j]*wr[j];
  DTB[((size_t)k*R + r)*128 + d] = softplusf(s);
}

// ---- gather concat(x,y) NCHW -> [R0=8192, 128] row-major ----------------------------
__global__ void k_gather_z(const float* __restrict__ x, const float* __restrict__ y,
                           float* __restrict__ Zt) {
  int i = blockIdx.x*blockDim.x + threadIdx.x;
  if (i >= 8192*128) return;
  int c = i & 127, r = i >> 7;
  int b = r >> 12; int hw = r & 4095;
  const float* src = (c < 64) ? x : y;
  Zt[i] = src[((size_t)(b*64 + (c & 63)))*4096 + hw];
}

// ---- LayerNorm over last dim (2-pass, thread per row) -------------------------------
__global__ void k_ln(const float* __restrict__ X, const float* __restrict__ w,
                     const float* __restrict__ b, float* __restrict__ Y, int R, int C) {
  int r = blockIdx.x*blockDim.x + threadIdx.x;
  if (r >= R) return;
  const float* xr = X + (size_t)r*C;
  float mu = 0.f, s2 = 0.f;
  for (int c = 0; c < C; ++c) { float v = xr[c]; mu += v; s2 += v*v; }
  mu /= (float)C;
  float var = s2/(float)C - mu*mu;
  float rs = rsqrtf(var + 1e-5f);
  float* yr = Y + (size_t)r*C;
  for (int c = 0; c < C; ++c) yr[c] = (xr[c]-mu)*rs*w[c] + b[c];
}

// ---- depthwise 3x3 conv + bias + SiLU, fused scatter into 4 scan directions ---------
__global__ void k_dwconv(const float* __restrict__ IP, const float* __restrict__ cw,
                         const float* __restrict__ cb, float* __restrict__ xs,
                         int B, int H, int W) {
  int L = H*W;
  int idx = blockIdx.x*blockDim.x + threadIdx.x;
  if (idx >= B*L*128) return;
  int d = idx & 127; int r = idx >> 7;
  int w0 = r % W; int h0 = (r / W) % H; int b = r / L;
  float s = cb[d];
  #pragma unroll
  for (int dh = -1; dh <= 1; ++dh)
    #pragma unroll
    for (int dw = -1; dw <= 1; ++dw) {
      int hh = h0 + dh, ww = w0 + dw;
      if (hh >= 0 && hh < H && ww >= 0 && ww < W)
        s += cw[d*9 + (dh+1)*3 + (dw+1)] * IP[((size_t)(b*L + hh*W + ww))*256 + d];
    }
  s = siluf(s);
  int l0 = h0*W + w0, l1 = w0*H + h0;
  size_t base = (size_t)b*L*128;
  size_t KBL  = (size_t)B*L*128;
  xs[0*KBL + base + (size_t)l0*128 + d]       = s;
  xs[1*KBL + base + (size_t)l1*128 + d]       = s;
  xs[2*KBL + base + (size_t)(L-1-l0)*128 + d] = s;
  xs[3*KBL + base + (size_t)(L-1-l1)*128 + d] = s;
}

// ---- selective scan: thread per (k,b,d) chain, LDS-staged B_t/C_t -------------------
// xs/dt/ys: [dir][B*L][128]; xd: [dir][B*L][48] (cols 4..19 = B, 20..35 = C)
__global__ void k_scan(const float* __restrict__ xs, const float* __restrict__ xd,
                       const float* __restrict__ dt, const float* __restrict__ A_log,
                       const float* __restrict__ Dp, float* __restrict__ ys,
                       int B, int L) {
  int kb = blockIdx.x; int k = kb / B; int b = kb % B;
  int d = threadIdx.x;                      // 0..127
  size_t roff = ((size_t)k*B + b) * (size_t)L;
  const float* xsk = xs + roff*128;
  const float* dtk = dt + roff*128;
  const float* xdk = xd + roff*48;
  float*       ysk = ys + roff*128;
  float Av[16];
  #pragma unroll
  for (int n = 0; n < 16; ++n) Av[n] = -__expf(A_log[((size_t)(k*128 + d))*16 + n]);
  float Dv = Dp[k*128 + d];
  float h[16];
  #pragma unroll
  for (int n = 0; n < 16; ++n) h[n] = 0.f;
  __shared__ float sB[16], sC[16];
  for (int t = 0; t < L; ++t) {
    if (d < 16)      sB[d]      = xdk[(size_t)t*48 + 4  + d];
    else if (d < 32) sC[d - 16] = xdk[(size_t)t*48 + 20 + (d - 16)];
    __syncthreads();
    float u   = xsk[(size_t)t*128 + d];
    float dtv = dtk[(size_t)t*128 + d];
    float du  = dtv * u;
    float y   = u * Dv;
    #pragma unroll
    for (int n = 0; n < 16; ++n) {
      h[n] = h[n]*__expf(dtv*Av[n]) + du*sB[n];
      y += h[n]*sC[n];
    }
    ysk[(size_t)t*128 + d] = y;
    __syncthreads();
  }
}

// ---- merge 4 directions + out_norm LN + SiLU(z) gate -> G[R,128] --------------------
__global__ void k_merge(const float* __restrict__ ys, const float* __restrict__ IP,
                        const float* __restrict__ nw, const float* __restrict__ nb,
                        float* __restrict__ G, int B, int H, int W) {
  int L = H*W;
  int r = blockIdx.x*blockDim.x + threadIdx.x;
  if (r >= B*L) return;
  int b = r / L, l = r % L;
  int w0 = l % W, h0 = l / W;
  int l1 = w0*H + h0;
  size_t KBL = (size_t)B*L*128;
  const float* y0 = ys + 0*KBL + ((size_t)(b*L + l))*128;
  const float* y1 = ys + 1*KBL + ((size_t)(b*L + l1))*128;
  const float* y2 = ys + 2*KBL + ((size_t)(b*L + (L-1-l)))*128;
  const float* y3 = ys + 3*KBL + ((size_t)(b*L + (L-1-l1)))*128;
  float mu = 0.f, s2 = 0.f;
  for (int d = 0; d < 128; ++d) {
    float v = y0[d] + y1[d] + y2[d] + y3[d];
    mu += v; s2 += v*v;
  }
  mu *= (1.f/128.f);
  float var = s2*(1.f/128.f) - mu*mu;
  float rs = rsqrtf(var + 1e-5f);
  const float* zg = IP + (size_t)r*256 + 128;
  float* gr = G + (size_t)r*128;
  for (int d = 0; d < 128; ++d) {
    float v = y0[d] + y1[d] + y2[d] + y3[d];
    float t = (v - mu)*rs*nw[d] + nb[d];
    gr[d] = t * siluf(zg[d]);
  }
}

// ---- downsample gather: 2x2/s2 patches -> [2048,256] --------------------------------
__global__ void k_gather_ds(const float* __restrict__ X2, float* __restrict__ DSG) {
  int i = blockIdx.x*blockDim.x + threadIdx.x;
  if (i >= 2048*256) return;
  int f = i & 255, r1 = i >> 8;
  int b = r1 >> 10; int hw = r1 & 1023; int h2 = hw >> 5; int w2 = hw & 31;
  int ci = f >> 2, kh = (f >> 1) & 1, kw = f & 1;
  int h = h2*2 + kh, w = w2*2 + kw;
  DSG[i] = X2[((size_t)(b*4096 + h*64 + w))*64 + ci];
}

// ---- global average pool over (h,w): [2048,64] -> [2,64] ----------------------------
__global__ void k_pool(const float* __restrict__ CLS, float* __restrict__ P) {
  int i = blockIdx.x*blockDim.x + threadIdx.x;
  if (i >= 128) return;
  int b = i >> 6, c = i & 63;
  float s = 0.f;
  for (int l = 0; l < 1024; ++l) s += CLS[((size_t)(b*1024 + l))*64 + c];
  P[i] = s * (1.f/1024.f);
}

// ---- classifier head: [2,64] x [1000,64]^T + bias -----------------------------------
__global__ void k_head(const float* __restrict__ P, const float* __restrict__ hw_,
                       const float* __restrict__ hb, float* __restrict__ out) {
  int i = blockIdx.x*blockDim.x + threadIdx.x;
  if (i >= 2000) return;
  int b = i / 1000, o = i % 1000;
  float s = hb[o];
  for (int c = 0; c < 64; ++c) s += P[b*64 + c] * hw_[o*64 + c];
  out[i] = s;
}

#define GEMM(ACT,HB,HA,KK,NT, Ap,Wp,bp,ap,Cp, M,N, lda,ldw,ldc,addld)                  \
  k_gemm<ACT,HB,HA,KK,NT><<<dim3((M)/16,(N)/(16*(NT))), dim3(32), 0, stream>>>(        \
      Ap, Wp, bp, ap, Cp, lda, ldw, ldc, addld)

extern "C" void kernel_launch(void* const* d_in, const int* in_sizes, int n_in,
                              void* d_out, int out_size, void* d_ws, size_t ws_size,
                              hipStream_t stream) {
  (void)in_sizes; (void)n_in; (void)out_size; (void)ws_size;
  const float* xin = (const float*)d_in[0];
  const float* yin = (const float*)d_in[1];
  auto P = [&](int i) { return (const float*)d_in[i]; };
  // d_in: 0 x, 1 y, 2 pe_w, 3 pe_b, 4 pe_ln_w, 5 pe_ln_b,
  //       6..24 blk0 (ln1_w,ln1_b,in_proj_w,conv_w,conv_b,x_proj_w,dt_w,dt_b,A_log,D,
  //                   out_norm_w,out_norm_b,out_proj_w,ln2_w,ln2_b,fc1_w,fc1_b,fc2_w,fc2_b)
  //       25 ds_w, 26 ds_b, 27 ds_ln_w, 28 ds_ln_b, 29..47 blk1,
  //       48 cls_ln_w, 49 cls_ln_b, 50 head_w, 51 head_b

  // ---- workspace carve (fp32) ----
  float* wsf = (float*)d_ws;
  size_t off = 0;
  auto A_ = [&](size_t n) { float* p = wsf + off; off += (n + 255) & ~(size_t)255; return p; };
  float* Zt   = A_((size_t)8192*128);
  float* E0   = A_((size_t)8192*64);
  float* Xres = A_((size_t)8192*64);
  float* LNB  = A_((size_t)8192*64);
  float* IP   = A_((size_t)8192*256);
  float* XS   = A_((size_t)4*8192*128);
  float* XD   = A_((size_t)4*8192*48);
  float* DTB  = A_((size_t)4*8192*128);
  float* YS   = A_((size_t)4*8192*128);
  float* G    = A_((size_t)8192*128);
  float* X1   = A_((size_t)8192*64);
  float* F1   = A_((size_t)8192*256);
  float* X2   = A_((size_t)8192*64);
  float* DSG  = A_((size_t)2048*256);
  float* PL   = A_(256);
  _Float16* H16 = (_Float16*)(wsf + off);
  size_t hoff = 0;
  auto H_ = [&](size_t n) { _Float16* p = H16 + hoff; hoff += (n + 255) & ~(size_t)255; return p; };
  _Float16* pe_h = H_(64*128);
  _Float16* ds_h = H_(64*256);
  struct BW { _Float16 *inp, *xp, *outp, *fc1, *fc2; } bw[2];
  for (int blk = 0; blk < 2; ++blk) {
    bw[blk].inp  = H_(256*64);
    bw[blk].xp   = H_((size_t)4*48*128);
    bw[blk].outp = H_(64*128);
    bw[blk].fc1  = H_(256*64);
    bw[blk].fc2  = H_(64*256);
  }

  // ---- per-call weight conversion to f16 [Npad,Kpad] ----
  auto cvt = [&](const float* src, _Float16* dst, int N, int K, int Npad, int Kpad) {
    int tot = Npad*Kpad;
    k_cvt<<<dim3((tot + 255)/256), dim3(256), 0, stream>>>(src, dst, N, K, Npad, Kpad);
  };
  cvt(P(2),  pe_h, 64, 128, 64, 128);
  cvt(P(25), ds_h, 64, 256, 64, 256);
  int bbase[2] = {6, 29};
  for (int blk = 0; blk < 2; ++blk) {
    int b0 = bbase[blk];
    cvt(P(b0 + 2),  bw[blk].inp,  256, 64, 256, 64);       // in_proj_w [256,64]
    for (int k = 0; k < 4; ++k)                            // x_proj_w [36,128] -> [48,128]
      cvt(P(b0 + 5) + (size_t)k*36*128, bw[blk].xp + (size_t)k*48*128, 36, 128, 48, 128);
    cvt(P(b0 + 12), bw[blk].outp, 64, 128, 64, 128);       // out_proj_w [64,128]
    cvt(P(b0 + 15), bw[blk].fc1,  256, 64, 256, 64);       // fc1_w [256,64]
    cvt(P(b0 + 17), bw[blk].fc2,  64, 256, 64, 256);       // fc2_w [64,256]
  }

  // ---- VSS block driver ----
  auto run_block = [&](int b0, const BW& w, float* XresP, float* XoutP, int H, int Wd) {
    int L = H*Wd; int R = BSZ*L;
    k_ln<<<dim3((R + 255)/256), dim3(256), 0, stream>>>(XresP, P(b0+0), P(b0+1), LNB, R, 64);
    GEMM(0,false,false,64,4, LNB, w.inp, nullptr, nullptr, IP, R, 256, 64, 64, 256, 0);
    {
      int tot = R*128;
      k_dwconv<<<dim3((tot + 255)/256), dim3(256), 0, stream>>>(IP, P(b0+3), P(b0+4), XS, BSZ, H, Wd);
    }
    for (int k = 0; k < 4; ++k)
      GEMM(0,false,false,128,3, XS + (size_t)k*R*128, w.xp + (size_t)k*48*128, nullptr, nullptr,
           XD + (size_t)k*R*48, R, 48, 128, 128, 48, 0);
    {
      int tot = 4*R*128;
      k_dt<<<dim3((tot + 255)/256), dim3(256), 0, stream>>>(XD, P(b0+6), P(b0+7), DTB, R);
    }
    k_scan<<<dim3(4*BSZ), dim3(128), 0, stream>>>(XS, XD, DTB, P(b0+8), P(b0+9), YS, BSZ, L);
    k_merge<<<dim3((R + 255)/256), dim3(256), 0, stream>>>(YS, IP, P(b0+10), P(b0+11), G, BSZ, H, Wd);
    GEMM(0,false,true,128,4, G, w.outp, nullptr, XresP, X1, R, 64, 128, 128, 64, 64);
    k_ln<<<dim3((R + 255)/256), dim3(256), 0, stream>>>(X1, P(b0+13), P(b0+14), LNB, R, 64);
    GEMM(1,true,false,64,4, LNB, w.fc1, P(b0+16), nullptr, F1, R, 256, 64, 64, 256, 0);
    GEMM(0,true,true,256,4, F1, w.fc2, P(b0+18), X1, XoutP, R, 64, 256, 256, 64, 64);
  };

  // ---- pipeline ----
  k_gather_z<<<dim3((8192*128 + 255)/256), dim3(256), 0, stream>>>(xin, yin, Zt);
  GEMM(0,true,false,128,4, Zt, pe_h, P(3), nullptr, E0, 8192, 64, 128, 128, 64, 0);   // patch embed
  k_ln<<<dim3((8192 + 255)/256), dim3(256), 0, stream>>>(E0, P(4), P(5), Xres, 8192, 64);
  run_block(6, bw[0], Xres, X2, 64, 64);                                              // VSS block 0
  k_gather_ds<<<dim3((2048*256 + 255)/256), dim3(256), 0, stream>>>(X2, DSG);
  GEMM(0,true,false,256,4, DSG, ds_h, P(26), nullptr, E0, 2048, 64, 256, 256, 64, 0); // downsample
  k_ln<<<dim3((2048 + 255)/256), dim3(256), 0, stream>>>(E0, P(27), P(28), Xres, 2048, 64);
  run_block(29, bw[1], Xres, X2, 32, 32);                                             // VSS block 1
  k_ln<<<dim3((2048 + 255)/256), dim3(256), 0, stream>>>(X2, P(48), P(49), LNB, 2048, 64);
  k_pool<<<dim3(1), dim3(128), 0, stream>>>(LNB, PL);
  k_head<<<dim3((2000 + 255)/256), dim3(256), 0, stream>>>(PL, P(50), P(51), (float*)d_out);
}
